// IDCM_34084860461113
// MI455X (gfx1250) — matrix operand order repaired and checked by reading, not compiled, not run on hardware
//
#include <hip/hip_runtime.h>
#include <math.h>

// ---------------- types / helpers ----------------
typedef __bf16 bf16_t;
typedef __attribute__((ext_vector_type(16))) __bf16 v16bf;
typedef __attribute__((ext_vector_type(8)))  __bf16 v8bf;
typedef __attribute__((ext_vector_type(8)))  float  v8f;

#define NB   32      // batch
#define LQ   30      // query len
#define LDOC 4000
#define NP   80      // chunks per doc
#define EXTL 64      // chunk length
#define OV   7
#define SN   8       // sampled chunks
#define DIM  768
#define FFD  3072
#define SEQL 94      // LQ + EXTL
#define NSEQ 256     // NB*SN
#define ROWS (NSEQ*SEQL)   // 24064
#define NHEAD 12
#define DH   64
#define NCHUNK (NB*NP)     // 2560

__device__ __constant__ float MU_c[11] =
  {1.0f,0.9f,0.7f,0.5f,0.3f,0.1f,-0.1f,-0.3f,-0.5f,-0.7f,-0.9f};

__device__ inline bf16_t f2bf(float f){
  unsigned u = __float_as_uint(f);
  u = u + 0x7FFFu + ((u >> 16) & 1u);       // RNE
  unsigned short h = (unsigned short)(u >> 16);
  return __builtin_bit_cast(__bf16, h);
}
__device__ inline float bf2f(bf16_t h){
  unsigned short s = __builtin_bit_cast(unsigned short, h);
  return __uint_as_float(((unsigned)s) << 16);
}
__device__ inline unsigned pack2bf(float a, float b){
  unsigned lo = (unsigned)__builtin_bit_cast(unsigned short, f2bf(a));
  unsigned hi = (unsigned)__builtin_bit_cast(unsigned short, f2bf(b));
  return lo | (hi << 16);
}

__device__ inline v16bf mk16(v8bf lo, v8bf hi){
  return __builtin_shufflevector(lo, hi, 0,1,2,3,4,5,6,7,8,9,10,11,12,13,14,15);
}
// A fragment: af[0..7]=row[k0+half*8 ..], af[8..15]=row[k0+16+half*8 ..]
__device__ inline v16bf afrag_load(const bf16_t* rowp, int k0, int half){
  v8bf lo = *reinterpret_cast<const v8bf*>(rowp + k0 + half * 8);
  v8bf hi = *reinterpret_cast<const v8bf*>(rowp + k0 + 16 + half * 8);
  return mk16(lo, hi);
}
// B fragment from N-major column: b[0..7]=col[k0+half*8 ..], b[8..15]=col[k0+16+half*8 ..]
__device__ inline v16bf bfrag_load(const bf16_t* colp, int k0, int half){
  v8bf lo = *reinterpret_cast<const v8bf*>(colp + k0 + half * 8);
  v8bf hi = *reinterpret_cast<const v8bf*>(colp + k0 + 16 + half * 8);
  return mk16(lo, hi);
}

// block-wide sum, result broadcast to all threads (blockDim.x multiple of 32)
__device__ inline float blockReduceSum(float v, float* red){
  for (int off = 16; off; off >>= 1) v += __shfl_down(v, off, 32);
  int wave = threadIdx.x >> 5, lane = threadIdx.x & 31;
  if (lane == 0) red[wave] = v;
  __syncthreads();
  int nw = blockDim.x >> 5;
  float t = (threadIdx.x < nw) ? red[threadIdx.x] : 0.f;
  if (wave == 0){
    for (int off = 16; off; off >>= 1) t += __shfl_down(t, off, 32);
    if (lane == 0) red[0] = t;
  }
  __syncthreads();
  float r = red[0];
  __syncthreads();
  return r;
}

__device__ inline v8f wmma_bf16(v16bf a, v16bf b, v8f c){
  return __builtin_amdgcn_wmma_f32_16x16x32_bf16(false, a, false, b, (short)0, c, false, false);
}

// ---------------- f32 (K x N) -> bf16 transposed (N x K) ----------------
__global__ __launch_bounds__(256) void cvt_t_kernel(const float* __restrict__ in,
                                                    bf16_t* __restrict__ out, int K, int N){
  __shared__ float tile[32][33];
  const float* src = in + (size_t)blockIdx.z * K * N;
  bf16_t* dst = out + (size_t)blockIdx.z * K * N;
  int k0 = blockIdx.y * 32, n0 = blockIdx.x * 32;
  int tx = threadIdx.x & 31, ty = threadIdx.x >> 5;   // 32 x 8
  for (int i = 0; i < 32; i += 8)
    tile[ty + i][tx] = src[(size_t)(k0 + ty + i) * N + n0 + tx];
  __syncthreads();
  for (int i = 0; i < 32; i += 8)
    dst[(size_t)(n0 + ty + i) * K + k0 + tx] = f2bf(tile[tx][ty + i]);
}

// ---------------- chunk id build ----------------
__global__ void chunk_build_kernel(const int* __restrict__ doc_ids,
                                   int* __restrict__ chunks, int* __restrict__ valid){
  int chunk = blockIdx.x;              // b*NP + p
  int b = chunk / NP, p = chunk % NP;
  int e = threadIdx.x;                 // 0..63
  __shared__ int f;
  if (e == 0) f = 0;
  __syncthreads();
  int j = p * 50 + e;                  // CHUNK=50
  int src = j - OV;
  int tok = (src >= 0 && src < (LDOC - 1)) ? doc_ids[b * LDOC + 1 + src] : 0;
  chunks[chunk * EXTL + e] = tok;
  if (e >= OV && e < EXTL - OV && tok != 0) atomicOr(&f, 1);
  __syncthreads();
  if (e == 0) valid[chunk] = f;
}

// ---------------- query embedding (padded rows for conv) ----------------
__global__ __launch_bounds__(256) void embq_kernel(
    const int* __restrict__ qids, const float* __restrict__ wemb, const float* __restrict__ pemb,
    const float* __restrict__ g, const float* __restrict__ be, bf16_t* __restrict__ qe){
  int row = blockIdx.x;                // 0..1025  (32*32 + 2 guard)
  int b = row / 32, l = row % 32;
  __shared__ float red[32];
  int tid = threadIdx.x;
  if (b >= NB || l >= LQ){
    for (int i = 0; i < 3; i++) qe[(size_t)row * DIM + tid + i * 256] = f2bf(0.f);
    return;
  }
  int id = qids[b * LQ + l];
  float v0 = wemb[(size_t)id * DIM + tid      ] + pemb[(size_t)l * DIM + tid      ];
  float v1 = wemb[(size_t)id * DIM + tid + 256] + pemb[(size_t)l * DIM + tid + 256];
  float v2 = wemb[(size_t)id * DIM + tid + 512] + pemb[(size_t)l * DIM + tid + 512];
  float mean = blockReduceSum(v0 + v1 + v2, red) * (1.f / DIM);
  float d0 = v0 - mean, d1 = v1 - mean, d2 = v2 - mean;
  float var = blockReduceSum(d0*d0 + d1*d1 + d2*d2, red) * (1.f / DIM);
  float rs = rsqrtf(var + 1e-12f);
  qe[(size_t)row*DIM + tid      ] = f2bf(d0*rs*g[tid      ] + be[tid      ]);
  qe[(size_t)row*DIM + tid + 256] = f2bf(d1*rs*g[tid + 256] + be[tid + 256]);
  qe[(size_t)row*DIM + tid + 512] = f2bf(d2*rs*g[tid + 512] + be[tid + 512]);
}

// ---------------- generic bf16 WMMA GEMM, double-buffered LDS ----------------
// C[M,N] = A[M,K] @ B[K,N], BT[n][k]; flags: 1=+bias, 2=+=outF(acc), (flags>>2)==2 -> gelu
__global__ __launch_bounds__(256) void gemm_bf16(
    const bf16_t* __restrict__ A, const bf16_t* __restrict__ BT,
    const float* __restrict__ bias, float* __restrict__ outF, bf16_t* __restrict__ outB,
    int M, int N, int K, int flags){
  __shared__ __align__(16) bf16_t As[2][64][40];
  __shared__ __align__(16) bf16_t Bs[2][128][40];
  int tid = threadIdx.x;
  int wave = tid >> 5, lane = tid & 31, half = lane >> 4, lsub = lane & 15;
  int mw = wave & 3, nw = wave >> 2;     // 4(M) x 2(N) waves
  int m0 = blockIdx.y * 64, n0 = blockIdx.x * 128;
  v8f acc[4];
  for (int i = 0; i < 4; i++) for (int j = 0; j < 8; j++) acc[i][j] = 0.f;
  int arow = tid >> 2, aks = (tid & 3) * 8;     // A stage: 64 rows x 32 k
  int bcol = tid >> 1, bks = (tid & 1) * 16;    // B stage: 128 cols x 32 k
  int gr = m0 + arow; if (gr >= M) gr = M - 1;
  const bf16_t* Aptr = A + (size_t)gr * K;
  const bf16_t* Bptr = BT + (size_t)(n0 + bcol) * K;
  // prologue: stage tile 0
  {
    uint4 ra  = *reinterpret_cast<const uint4*>(Aptr + aks);
    const uint4* bp = reinterpret_cast<const uint4*>(Bptr + bks);
    uint4 rb0 = bp[0], rb1 = bp[1];
    *reinterpret_cast<uint4*>(&As[0][arow][aks])     = ra;
    *reinterpret_cast<uint4*>(&Bs[0][bcol][bks])     = rb0;
    *reinterpret_cast<uint4*>(&Bs[0][bcol][bks + 8]) = rb1;
  }
  __syncthreads();
  int cur = 0;
  int ar = mw * 16 + lsub;
  int col0 = nw * 64 + lsub;
  for (int k0 = 32; k0 < K; k0 += 32){
    // issue next tile's global loads
    uint4 ra  = *reinterpret_cast<const uint4*>(Aptr + k0 + aks);
    const uint4* bp = reinterpret_cast<const uint4*>(Bptr + k0 + bks);
    uint4 rb0 = bp[0], rb1 = bp[1];
    // compute current tile (all fragment loads issued before the wmmas)
    v16bf af = afrag_load(&As[cur][ar][0], 0, half);
    v16bf b0 = bfrag_load(&Bs[cur][col0     ][0], 0, half);
    v16bf b1 = bfrag_load(&Bs[cur][col0 + 16][0], 0, half);
    v16bf b2 = bfrag_load(&Bs[cur][col0 + 32][0], 0, half);
    v16bf b3 = bfrag_load(&Bs[cur][col0 + 48][0], 0, half);
    acc[0] = wmma_bf16(af, b0, acc[0]);
    acc[1] = wmma_bf16(af, b1, acc[1]);
    acc[2] = wmma_bf16(af, b2, acc[2]);
    acc[3] = wmma_bf16(af, b3, acc[3]);
    // store next tile into alternate buffer
    int nxt = cur ^ 1;
    *reinterpret_cast<uint4*>(&As[nxt][arow][aks])     = ra;
    *reinterpret_cast<uint4*>(&Bs[nxt][bcol][bks])     = rb0;
    *reinterpret_cast<uint4*>(&Bs[nxt][bcol][bks + 8]) = rb1;
    __syncthreads();
    cur = nxt;
  }
  // epilogue compute (last tile)
  {
    v16bf af = afrag_load(&As[cur][ar][0], 0, half);
    v16bf b0 = bfrag_load(&Bs[cur][col0     ][0], 0, half);
    v16bf b1 = bfrag_load(&Bs[cur][col0 + 16][0], 0, half);
    v16bf b2 = bfrag_load(&Bs[cur][col0 + 32][0], 0, half);
    v16bf b3 = bfrag_load(&Bs[cur][col0 + 48][0], 0, half);
    acc[0] = wmma_bf16(af, b0, acc[0]);
    acc[1] = wmma_bf16(af, b1, acc[1]);
    acc[2] = wmma_bf16(af, b2, acc[2]);
    acc[3] = wmma_bf16(af, b3, acc[3]);
  }
  for (int nt = 0; nt < 4; nt++){
    int col = n0 + nw * 64 + nt * 16 + lsub;
    for (int j = 0; j < 8; j++){
      int row = m0 + mw * 16 + half * 8 + j;
      if (row < M){
        float v = acc[nt][j];
        if (flags & 1) v += bias[col];
        size_t off = (size_t)row * N + col;
        if (flags & 2) v += outF[off];
        if (((flags >> 2) & 3) == 2) v = 0.5f * v * (1.f + erff(v * 0.70710678118f));
        if (outF) outF[off] = v;
        if (outB) outB[off] = f2bf(v);
      }
    }
  }
}

// ---------------- relu + L2 normalize query context ----------------
__global__ __launch_bounds__(256) void qnorm_kernel(const float* __restrict__ yq,
                                                    bf16_t* __restrict__ qctx){
  int row = blockIdx.x;               // 0..1023  (b*32 + l)
  int l = row % 32;
  __shared__ float red[32];
  int tid = threadIdx.x;
  if (l >= LQ){
    for (int i = 0; i < 3; i++) qctx[(size_t)row * DIM + tid + i * 256] = f2bf(0.f);
    return;
  }
  float v[3], ss = 0.f;
  for (int i = 0; i < 3; i++){
    float y = yq[(size_t)row * DIM + tid + i * 256];
    y = y > 0.f ? y : 0.f;
    v[i] = y; ss += y * y;
  }
  float nrm = sqrtf(blockReduceSum(ss, red));
  float sc = 1.f / fmaxf(nrm, 1e-12f);
  for (int i = 0; i < 3; i++) qctx[(size_t)row * DIM + tid + i * 256] = f2bf(v[i] * sc);
}

// ---------------- fused chunk scorer (emb+LN -> conv WMMA -> relu/norm -> cos WMMA -> pool)
__global__ __launch_bounds__(256) void chunk_score_kernel(
    const int* __restrict__ chunks, const bf16_t* __restrict__ qctx,
    const bf16_t* __restrict__ convBT, const float* __restrict__ convb,
    const float* __restrict__ wemb, const float* __restrict__ pemb,
    const float* __restrict__ g, const float* __restrict__ be,
    float* __restrict__ kacc){
  int blk = blockIdx.x;
  int chunk = blk >> 2, quarter = blk & 3;
  int b = chunk / NP;
  __shared__ __align__(16) bf16_t Ae[18][776];   // emb rows e0..e0+17
  __shared__ __align__(16) bf16_t Yt[16][776];   // relu(conv) rows (local e)
  __shared__ float red[32];
  __shared__ float nrm[16];
  __shared__ float cosL[32][17];
  __shared__ float kpart[30][11];
  __shared__ float dm[16];
  int tid = threadIdx.x;
  if (tid < 16){
    nrm[tid] = 0.f;
    dm[tid] = (chunks[chunk * EXTL + quarter * 16 + tid] != 0) ? 1.f : 0.f;
  }
  for (int i = tid; i < 330; i += 256) kpart[i / 11][i % 11] = 0.f;
  // --- embed + LN 18 rows ---
  int e0 = quarter * 16;
  for (int r = 0; r < 18; r++){
    int m = e0 + r;
    if (m >= EXTL){
      for (int i = 0; i < 3; i++) Ae[r][tid + i * 256] = f2bf(0.f);
      continue;
    }
    int id = chunks[chunk * EXTL + m];
    float v0 = wemb[(size_t)id * DIM + tid      ] + pemb[(size_t)m * DIM + tid      ];
    float v1 = wemb[(size_t)id * DIM + tid + 256] + pemb[(size_t)m * DIM + tid + 256];
    float v2 = wemb[(size_t)id * DIM + tid + 512] + pemb[(size_t)m * DIM + tid + 512];
    float mean = blockReduceSum(v0 + v1 + v2, red) * (1.f / DIM);
    float d0 = v0 - mean, d1 = v1 - mean, d2 = v2 - mean;
    float var = blockReduceSum(d0*d0 + d1*d1 + d2*d2, red) * (1.f / DIM);
    float rs = rsqrtf(var + 1e-12f);
    Ae[r][tid      ] = f2bf(d0 * rs * g[tid      ] + be[tid      ]);
    Ae[r][tid + 256] = f2bf(d1 * rs * g[tid + 256] + be[tid + 256]);
    Ae[r][tid + 512] = f2bf(d2 * rs * g[tid + 512] + be[tid + 512]);
  }
  __syncthreads();
  // --- conv GEMM: Y[16,768] = sum_tap Ae[e+tap] @ convW[tap], weights N-major from L2 ---
  int wave = tid >> 5, lane = tid & 31, half = lane >> 4, lsub = lane & 15;
  int ncol0 = wave * 96;
  for (int nt = 0; nt < 6; nt++){
    v8f acc; for (int j = 0; j < 8; j++) acc[j] = 0.f;
    int col = ncol0 + nt * 16 + lsub;
    for (int tap = 0; tap < 3; tap++){
      const bf16_t* colp = convBT + ((size_t)tap * DIM + col) * DIM;
      int ar = lsub + tap;
      #pragma unroll 4
      for (int k0 = 0; k0 < DIM; k0 += 32){
        v16bf af  = afrag_load(&Ae[ar][0], k0, half);
        v16bf bfv = bfrag_load(colp, k0, half);
        acc = wmma_bf16(af, bfv, acc);
      }
    }
    float bia = convb[col];
    for (int j = 0; j < 8; j++){
      int er = half * 8 + j;
      float y = acc[j] + bia;
      y = y > 0.f ? y : 0.f;
      atomicAdd(&nrm[er], y * y);
      Yt[er][col] = f2bf(y);
    }
  }
  __syncthreads();
  // --- cos GEMM: qctx[b](32x768) @ Yt^T(768x16) by waves 0,1 ---
  if (wave < 2){
    v8f acc; for (int j = 0; j < 8; j++) acc[j] = 0.f;
    const bf16_t* Aq = qctx + (size_t)b * 32 * DIM;
    int ar = wave * 16 + lsub;
    #pragma unroll 4
    for (int k0 = 0; k0 < DIM; k0 += 32){
      v16bf af  = afrag_load(Aq + (size_t)ar * DIM, k0, half);
      v16bf bfv = bfrag_load(&Yt[lsub][0], k0, half);
      acc = wmma_bf16(af, bfv, acc);
    }
    for (int j = 0; j < 8; j++) cosL[wave * 16 + half * 8 + j][lsub] = acc[j];
  }
  __syncthreads();
  // --- kernel pooling ---
  for (int idx = tid; idx < 480; idx += 256){
    int q = idx >> 4, e = idx & 15;
    if (dm[e] != 0.f){
      float c = cosL[q][e] / fmaxf(sqrtf(nrm[e]), 1e-12f);
      #pragma unroll
      for (int mu = 0; mu < 11; mu++){
        float d = c - MU_c[mu];
        atomicAdd(&kpart[q][mu], __expf(-d * d * 50.0f));   // 1/(2*0.1^2)=50
      }
    }
  }
  __syncthreads();
  for (int idx = tid; idx < 330; idx += 256){
    int q = idx / 11, mu = idx % 11;
    atomicAdd(&kacc[((size_t)chunk * LQ + q) * 11 + mu], kpart[q][mu]);
  }
}

// ---------------- samp ----------------
__global__ void samp_kernel(const float* __restrict__ kacc, const float* __restrict__ alpha,
                            const float* __restrict__ binw, const float* __restrict__ binwb,
                            const int* __restrict__ qmask, const int* __restrict__ valid,
                            float* __restrict__ samp){
  int idx = blockIdx.x * blockDim.x + threadIdx.x;
  if (idx >= NCHUNK) return;
  int b = idx / NP;
  float s = 0.f;
  for (int q = 0; q < LQ; q++){
    float qm = qmask[b * LQ + q] ? 1.f : 0.f;
    for (int mu = 0; mu < 11; mu++){
      float kr = logf(fmaxf(kacc[((size_t)idx * LQ + q) * 11 + mu] * alpha[mu], 1e-4f)) * qm;
      s += kr * binw[mu];
    }
  }
  s += binwb[0];
  if (!valid[idx]) s = 0.f;
  if (s == 0.f) s = -9000.f;
  samp[idx] = s;
}

// ---------------- stable top-8 ----------------
__global__ void topk_kernel(const float* __restrict__ samp, int* __restrict__ topidx){
  int b = threadIdx.x;
  if (b >= NB) return;
  bool used[NP];
  for (int p = 0; p < NP; p++) used[p] = false;
  for (int s = 0; s < SN; s++){
    float best = -1e30f; int bi = 0;
    for (int p = 0; p < NP; p++){
      float v = samp[b * NP + p];
      if (!used[p] && v > best){ best = v; bi = p; }
    }
    used[bi] = true;
    topidx[b * SN + s] = bi;
  }
}

// ---------------- BERT input embedding + LN + mask ----------------
__global__ __launch_bounds__(256) void emb_bert_kernel(
    const int* __restrict__ qids, const int* __restrict__ qmask, const int* __restrict__ chunks,
    const int* __restrict__ topidx, const float* __restrict__ wemb, const float* __restrict__ pemb,
    const float* __restrict__ g, const float* __restrict__ be,
    float* __restrict__ xf, bf16_t* __restrict__ xb, int* __restrict__ am){
  int row = blockIdx.x;             // 0..ROWS-1
  int seq = row / SEQL, pos = row % SEQL;
  int b = seq / SN, s = seq % SN;
  int id, m;
  if (pos < LQ){ id = qids[b * LQ + pos]; m = (qmask[b * LQ + pos] != 0); }
  else { int p = topidx[b * SN + s]; id = chunks[(b * NP + p) * EXTL + (pos - LQ)]; m = (id != 0); }
  if (threadIdx.x == 0) am[row] = m;
  __shared__ float red[32];
  int tid = threadIdx.x;
  float v0 = wemb[(size_t)id * DIM + tid      ] + pemb[(size_t)pos * DIM + tid      ];
  float v1 = wemb[(size_t)id * DIM + tid + 256] + pemb[(size_t)pos * DIM + tid + 256];
  float v2 = wemb[(size_t)id * DIM + tid + 512] + pemb[(size_t)pos * DIM + tid + 512];
  float mean = blockReduceSum(v0 + v1 + v2, red) * (1.f / DIM);
  float d0 = v0 - mean, d1 = v1 - mean, d2 = v2 - mean;
  float var = blockReduceSum(d0*d0 + d1*d1 + d2*d2, red) * (1.f / DIM);
  float rs = rsqrtf(var + 1e-12f);
  float y0 = d0*rs*g[tid] + be[tid], y1 = d1*rs*g[tid+256] + be[tid+256], y2 = d2*rs*g[tid+512] + be[tid+512];
  size_t o = (size_t)row * DIM + tid;
  xf[o] = y0; xf[o + 256] = y1; xf[o + 512] = y2;
  xb[o] = f2bf(y0); xb[o + 256] = f2bf(y1); xb[o + 512] = f2bf(y2);
}

// ---------------- attention: one (seq, head) per block, packed bf16 pairs ----------------
__global__ __launch_bounds__(128) void attn_kernel(
    const bf16_t* __restrict__ q, const bf16_t* __restrict__ k, const bf16_t* __restrict__ v,
    const int* __restrict__ am, bf16_t* __restrict__ ctx){
  int blk = blockIdx.x;
  int seq = blk / NHEAD, h = blk % NHEAD;
  __shared__ unsigned ksu[SEQL][DH / 2];
  __shared__ unsigned vsu[SEQL][DH / 2];
  __shared__ float sc[SEQL][96];
  size_t base = (size_t)seq * SEQL * DIM + h * DH;   // element offset (even)
  for (int idx = threadIdx.x; idx < SEQL * (DH / 2); idx += 128){
    int r = idx >> 5, c = idx & 31;
    const unsigned* kp = reinterpret_cast<const unsigned*>(k + base + (size_t)r * DIM);
    const unsigned* vp = reinterpret_cast<const unsigned*>(v + base + (size_t)r * DIM);
    ksu[r][c] = kp[c];
    vsu[r][c] = vp[c];
  }
  __syncthreads();
  int t = threadIdx.x;
  if (t < SEQL){
    float qr[DH];
    const uint4* qp = reinterpret_cast<const uint4*>(q + base + (size_t)t * DIM);
    for (int c8 = 0; c8 < 8; c8++){
      uint4 u4 = qp[c8];
      unsigned ua[4] = {u4.x, u4.y, u4.z, u4.w};
      for (int j = 0; j < 4; j++){
        qr[c8 * 8 + 2 * j    ] = __uint_as_float(ua[j] << 16);
        qr[c8 * 8 + 2 * j + 1] = __uint_as_float(ua[j] & 0xFFFF0000u);
      }
    }
    float mx = -1e30f;
    for (int j = 0; j < SEQL; j++){
      float s = 0.f;
      for (int c = 0; c < DH / 2; c++){
        unsigned u = ksu[j][c];
        s += qr[2 * c    ] * __uint_as_float(u << 16);
        s += qr[2 * c + 1] * __uint_as_float(u & 0xFFFF0000u);
      }
      s *= 0.125f;                         // 1/sqrt(64)
      if (!am[seq * SEQL + j]) s = -10000.0f;
      sc[t][j] = s;
      if (s > mx) mx = s;
    }
    float sum = 0.f;
    for (int j = 0; j < SEQL; j++){ float e = __expf(sc[t][j] - mx); sc[t][j] = e; sum += e; }
    float inv = 1.0f / sum;
    unsigned* ctxu = reinterpret_cast<unsigned*>(ctx + base + (size_t)t * DIM);
    for (int c = 0; c < DH / 2; c++){
      float o0 = 0.f, o1 = 0.f;
      for (int j = 0; j < SEQL; j++){
        unsigned u = vsu[j][c];
        float pj = sc[t][j];
        o0 += pj * __uint_as_float(u << 16);
        o1 += pj * __uint_as_float(u & 0xFFFF0000u);
      }
      ctxu[c] = pack2bf(o0 * inv, o1 * inv);
    }
  }
}

// ---------------- residual add + layernorm (writes f32 + bf16) ----------------
__global__ __launch_bounds__(256) void addln_kernel(
    float* __restrict__ xf, const float* __restrict__ t0,
    const float* __restrict__ g, const float* __restrict__ be, bf16_t* __restrict__ xb){
  int row = blockIdx.x;
  __shared__ float red[32];
  int tid = threadIdx.x;
  size_t o = (size_t)row * DIM + tid;
  float v0 = xf[o] + t0[o], v1 = xf[o + 256] + t0[o + 256], v2 = xf[o + 512] + t0[o + 512];
  float mean = blockReduceSum(v0 + v1 + v2, red) * (1.f / DIM);
  float d0 = v0 - mean, d1 = v1 - mean, d2 = v2 - mean;
  float var = blockReduceSum(d0*d0 + d1*d1 + d2*d2, red) * (1.f / DIM);
  float rs = rsqrtf(var + 1e-12f);
  float y0 = d0*rs*g[tid] + be[tid], y1 = d1*rs*g[tid+256] + be[tid+256], y2 = d2*rs*g[tid+512] + be[tid+512];
  xf[o] = y0; xf[o + 256] = y1; xf[o + 512] = y2;
  xb[o] = f2bf(y0); xb[o + 256] = f2bf(y1); xb[o + 512] = f2bf(y2);
}

// ---------------- CLS score ----------------
__global__ __launch_bounds__(256) void cls_kernel(
    const float* __restrict__ xf, const float* __restrict__ clsw, const float* __restrict__ clsb,
    const int* __restrict__ topidx, const int* __restrict__ valid, float* __restrict__ bsc){
  int seq = blockIdx.x;
  __shared__ float red[32];
  float s = 0.f;
  for (int c = threadIdx.x; c < DIM; c += 256)
    s += xf[(size_t)seq * SEQL * DIM + c] * clsw[c];
  s = blockReduceSum(s, red);
  if (threadIdx.x == 0){
    float v = s + clsb[0];
    int b = seq / SN, k = seq % SN;
    if (!valid[b * NP + topidx[b * SN + k]]) v = 0.f;
    bsc[seq] = v;
  }
}

// ---------------- finale: sort desc, top-3 dot ----------------
__global__ void final_kernel(const float* __restrict__ bsc, const float* __restrict__ topkw,
                             float* __restrict__ out){
  int b = threadIdx.x;
  if (b >= NB) return;
  float v[SN];
  for (int i = 0; i < SN; i++){
    float x = bsc[b * SN + i];
    if (x == 0.f) x = -9000.f;
    v[i] = x;
  }
  for (int i = 0; i < SN; i++)
    for (int j = i + 1; j < SN; j++)
      if (v[j] > v[i]){ float t = v[i]; v[i] = v[j]; v[j] = t; }
  float o = 0.f;
  for (int t = 0; t < 3; t++){
    float x = v[t];
    if (x <= -8900.f) x = 0.f;
    o += x * topkw[t];
  }
  out[b] = o;
}

// ---------------- launcher ----------------
static inline size_t alignup(size_t x){ return (x + 255) & ~(size_t)255; }

extern "C" void kernel_launch(void* const* d_in, const int* in_sizes, int n_in,
                              void* d_out, int out_size, void* d_ws, size_t ws_size,
                              hipStream_t stream){
  const int*   query_ids  = (const int*)  d_in[0];
  const int*   query_mask = (const int*)  d_in[1];
  const int*   doc_ids    = (const int*)  d_in[2];
  const float* word_emb   = (const float*)d_in[3];
  const float* pos_emb    = (const float*)d_in[4];
  const float* emb_g      = (const float*)d_in[5];
  const float* emb_b      = (const float*)d_in[6];
  const float* conv_w     = (const float*)d_in[7];
  const float* conv_b     = (const float*)d_in[8];
  const float* alpha      = (const float*)d_in[9];
  const float* binw_w     = (const float*)d_in[10];
  const float* binw_b     = (const float*)d_in[11];
  const float* cls_w      = (const float*)d_in[12];
  const float* cls_b      = (const float*)d_in[13];
  const float* topk_w     = (const float*)d_in[14];
  const float* Wq = (const float*)d_in[15]; const float* bq = (const float*)d_in[16];
  const float* Wk = (const float*)d_in[17]; const float* bk = (const float*)d_in[18];
  const float* Wv = (const float*)d_in[19]; const float* bv = (const float*)d_in[20];
  const float* Wo = (const float*)d_in[21]; const float* bo = (const float*)d_in[22];
  const float* ln1g = (const float*)d_in[23]; const float* ln1b = (const float*)d_in[24];
  const float* W1 = (const float*)d_in[25]; const float* bf1 = (const float*)d_in[26];
  const float* W2 = (const float*)d_in[27]; const float* bf2 = (const float*)d_in[28];
  const float* ln2g = (const float*)d_in[29]; const float* ln2b = (const float*)d_in[30];
  float* out = (float*)d_out;

  char* w = (char*)d_ws;
  size_t off = 0;
  auto alloc = [&](size_t bytes){ void* p = w + off; off = alignup(off + bytes); return p; };

  int*    chunks = (int*)   alloc((size_t)NCHUNK * EXTL * 4);
  int*    valid  = (int*)   alloc((size_t)NCHUNK * 4);
  float*  kacc   = (float*) alloc((size_t)NCHUNK * LQ * 11 * 4);
  float*  samp   = (float*) alloc((size_t)NCHUNK * 4);
  int*    topidx = (int*)   alloc((size_t)NB * SN * 4);
  bf16_t* qe     = (bf16_t*)alloc((size_t)1026 * DIM * 2);
  float*  yq     = (float*) alloc((size_t)1024 * DIM * 4);
  bf16_t* qctx   = (bf16_t*)alloc((size_t)1024 * DIM * 2);
  bf16_t* convBT = (bf16_t*)alloc((size_t)3 * DIM * DIM * 2);
  bf16_t* WqT    = (bf16_t*)alloc((size_t)4 * DIM * DIM * 2);
  bf16_t* WkT    = (bf16_t*)alloc((size_t)4 * DIM * DIM * 2);
  bf16_t* WvT    = (bf16_t*)alloc((size_t)4 * DIM * DIM * 2);
  bf16_t* WoT    = (bf16_t*)alloc((size_t)4 * DIM * DIM * 2);
  bf16_t* W1T    = (bf16_t*)alloc((size_t)4 * DIM * FFD * 2);
  bf16_t* W2T    = (bf16_t*)alloc((size_t)4 * FFD * DIM * 2);
  float*  xf     = (float*) alloc((size_t)ROWS * DIM * 4);
  bf16_t* xb     = (bf16_t*)alloc((size_t)ROWS * DIM * 2);
  float*  t0     = (float*) alloc((size_t)ROWS * DIM * 4);
  int*    am     = (int*)   alloc((size_t)ROWS * 4);
  char*   big    = (char*)  alloc((size_t)ROWS * FFD * 2);   // 147.8 MB overlay
  float*  bsc    = (float*) alloc((size_t)NSEQ * 4);

  size_t actB = (size_t)ROWS * DIM * 2;
  bf16_t* qb   = (bf16_t*)(big);
  bf16_t* kb_  = (bf16_t*)(big + actB);
  bf16_t* vb   = (bf16_t*)(big + 2 * actB);
  bf16_t* ctxb = (bf16_t*)(big + 3 * actB);
  bf16_t* hb   = (bf16_t*)(big);                              // phase-B overlay

  (void)hipMemsetAsync(kacc, 0, (size_t)NCHUNK * LQ * 11 * 4, stream);

  // transpose+convert weights: in (K x N f32) -> out (N x K bf16), batched over layers
  auto cvtT = [&](const float* src, bf16_t* dst, int K, int N, int layers){
    cvt_t_kernel<<<dim3(N / 32, K / 32, layers), 256, 0, stream>>>(src, dst, K, N);
  };
  cvtT(conv_w, convBT, DIM, DIM, 3);
  cvtT(Wq, WqT, DIM, DIM, 4);
  cvtT(Wk, WkT, DIM, DIM, 4);
  cvtT(Wv, WvT, DIM, DIM, 4);
  cvtT(Wo, WoT, DIM, DIM, 4);
  cvtT(W1, W1T, DIM, FFD, 4);
  cvtT(W2, W2T, FFD, DIM, 4);

  chunk_build_kernel<<<NCHUNK, 64, 0, stream>>>(doc_ids, chunks, valid);
  embq_kernel<<<1026, 256, 0, stream>>>(query_ids, word_emb, pos_emb, emb_g, emb_b, qe);

  auto gemm = [&](const bf16_t* A, const bf16_t* Bt, const float* bias,
                  float* oF, bf16_t* oB, int M, int N, int K, int flags){
    gemm_bf16<<<dim3(N / 128, M / 64), 256, 0, stream>>>(A, Bt, bias, oF, oB, M, N, K, flags);
  };

  // query conv: 3 taps accumulate (A row-shifted views of qe)
  gemm(qe,           convBT,                      conv_b,  yq, nullptr, 1024, DIM, DIM, 1);
  gemm(qe + DIM,     convBT + (size_t)DIM * DIM,  nullptr, yq, nullptr, 1024, DIM, DIM, 2);
  gemm(qe + 2 * DIM, convBT + (size_t)2*DIM*DIM,  nullptr, yq, nullptr, 1024, DIM, DIM, 2);
  qnorm_kernel<<<1024, 256, 0, stream>>>(yq, qctx);

  chunk_score_kernel<<<NCHUNK * 4, 256, 0, stream>>>(chunks, qctx, convBT, conv_b,
                                                     word_emb, pos_emb, emb_g, emb_b, kacc);
  samp_kernel<<<(NCHUNK + 255) / 256, 256, 0, stream>>>(kacc, alpha, binw_w, binw_b,
                                                        query_mask, valid, samp);
  topk_kernel<<<1, 32, 0, stream>>>(samp, topidx);

  emb_bert_kernel<<<ROWS, 256, 0, stream>>>(query_ids, query_mask, chunks, topidx,
                                            word_emb, pos_emb, emb_g, emb_b, xf, xb, am);

  for (int l = 0; l < 4; l++){
    size_t wO = (size_t)l * DIM * DIM, wF = (size_t)l * DIM * FFD;
    gemm(xb, WqT + wO, bq + l * DIM, nullptr, qb,  ROWS, DIM, DIM, 1);
    gemm(xb, WkT + wO, bk + l * DIM, nullptr, kb_, ROWS, DIM, DIM, 1);
    gemm(xb, WvT + wO, bv + l * DIM, nullptr, vb,  ROWS, DIM, DIM, 1);
    attn_kernel<<<NSEQ * NHEAD, 128, 0, stream>>>(qb, kb_, vb, am, ctxb);
    gemm(ctxb, WoT + wO, bo + l * DIM, t0, nullptr, ROWS, DIM, DIM, 1);
    addln_kernel<<<ROWS, 256, 0, stream>>>(xf, t0, ln1g + l * DIM, ln1b + l * DIM, xb);
    gemm(xb, W1T + wF, bf1 + l * FFD, nullptr, hb, ROWS, FFD, DIM, 1 | 8);   // gelu -> bf16
    gemm(hb, W2T + wF, bf2 + l * DIM, t0, nullptr, ROWS, DIM, FFD, 1);
    addln_kernel<<<ROWS, 256, 0, stream>>>(xf, t0, ln2g + l * DIM, ln2b + l * DIM, xb);
  }

  cls_kernel<<<NSEQ, 256, 0, stream>>>(xf, cls_w, cls_b, topidx, valid, bsc);
  final_kernel<<<1, 32, 0, stream>>>(bsc, topk_w, out);
}
